// TransformerPointerNet_78649441124902
// MI455X (gfx1250) — compile-verified, matrix operand order checked by reading, not run on hardware
//
#include <hip/hip_runtime.h>
#include <hip/hip_bf16.h>
#include <math.h>

typedef _Float16 f16;
typedef __attribute__((ext_vector_type(16))) _Float16 v16h;
typedef __attribute__((ext_vector_type(8)))  float v8f;
typedef __attribute__((ext_vector_type(4)))  float f32x4;

// Model constants
#define NHh 8
#define Hdim 128
#define Bb 64
#define Nn 32
#define Tq 33

// Force the GEMM fragment loads onto the GLOBAL vmem path (global_load_b128)
// instead of FLAT (which also ties up DScnt / the LDS arbitration path).
#define GAS __attribute__((address_space(1)))
__device__ __forceinline__ v16h tpn_load_frag(const f16* p) {
  const GAS f32x4* q = (const GAS f32x4*)p;   // 16B-aligned, K-swizzled layout
  v16h f;
  ((f32x4*)&f)[0] = q[0];    // element e=0..7  : K = e + 8*gg
  ((f32x4*)&f)[1] = q[2];    // element e=8..15 : K = e + 8 + 8*gg
  return f;
}

// ---------------------------------------------------------------------------
// Generic batched WMMA GEMM:  C[M,N] = act(A[M,K] @ W[N,K]^T + bias)
// A, W are f16 row-major with K contiguous (K multiple of 32, buffers padded
// with zeros so no load guards are needed -> EXEC stays all-ones for WMMA).
// Two-level batch strides (b1,b2) cover per-(batch,head) attention GEMMs.
// NT = number of active 16-wide n-subtiles (compile-time: branch-free K loop).
// ---------------------------------------------------------------------------
struct GemmArgs {
  const f16* A; long long sA1, sA2; int lda;
  const f16* W; long long sW1, sW2; int ldw;
  const float* bias;
  float* C;  long long sC1, sC2; int ldc;
  f16*  C16; long long sD1, sD2; int ldc16;
  int M, N, K, NB2, act;
};

template <int NT>
__global__ __launch_bounds__(128) void tpn_gemm_wmma(GemmArgs g) {
  const int lane = threadIdx.x & 31;
  const int wave = threadIdx.x >> 5;
  const int mt = blockIdx.x * 4 + wave;
  if (mt * 16 >= g.M) return;                       // wave-uniform
  const int b1 = (int)blockIdx.z / g.NB2;
  const int b2 = (int)blockIdx.z % g.NB2;
  const f16* A = g.A + b1 * g.sA1 + b2 * g.sA2;
  const f16* W = g.W + b1 * g.sW1 + b2 * g.sW2;
  const int r  = lane & 15;                          // A row / W row / C col
  const int gg = lane >> 4;                          // K-half select
  const int nbase = blockIdx.y * 64;

  v8f acc[NT];
#pragma unroll
  for (int j = 0; j < NT; ++j)
#pragma unroll
    for (int v = 0; v < 8; ++v) acc[j][v] = 0.f;

  const f16* Arow = A + (long long)(mt * 16 + r) * g.lda + 8 * gg;
  const f16* Wrow[NT];
#pragma unroll
  for (int j = 0; j < NT; ++j)
    Wrow[j] = W + (long long)(nbase + j * 16 + r) * g.ldw + 8 * gg;

  for (int k0 = 0; k0 < g.K; k0 += 32) {
    // 16-bit fragment: element e holds K = e + 8*((e>>3)+gg)  (ISA 7.12.2)
    v16h a = tpn_load_frag(Arow);
    Arow += 32;
#pragma unroll
    for (int j = 0; j < NT; ++j) {
      v16h w = tpn_load_frag(Wrow[j]);
      Wrow[j] += 32;
      acc[j] = __builtin_amdgcn_wmma_f32_16x16x32_f16(
          false, a, false, w, (short)0, acc[j], false, false);
    }
  }

  float* C = g.C  ? g.C  + b1 * g.sC1 + b2 * g.sC2 : nullptr;
  f16*   D = g.C16 ? g.C16 + b1 * g.sD1 + b2 * g.sD2 : nullptr;
#pragma unroll
  for (int j = 0; j < NT; ++j) {
    const int n = nbase + j * 16 + r;
    if (n >= g.N) continue;
    const float bv = g.bias ? g.bias[n] : 0.f;
#pragma unroll
    for (int v = 0; v < 8; ++v) {
      const int m = mt * 16 + v + 8 * gg;            // C layout: VGPR v -> M
      if (m >= g.M) continue;
      float val = acc[j][v] + bv;
      if (g.act == 1) val = fmaxf(val, 0.f);
      if (C) C[(long long)m * g.ldc + n] = val;
      if (D) D[(long long)m * g.ldc16 + n] = (f16)val;
    }
  }
}

// ------------------------- small support kernels ---------------------------
__global__ void tpn_cvt_f32_f16(const float* src, f16* dst, int R, int Csrc,
                                int ldd, int tr) {
  int i = blockIdx.x * blockDim.x + threadIdx.x;
  if (i >= R * ldd) return;
  int rr = i / ldd, c = i - rr * ldd;
  float v = 0.f;
  if (c < Csrc) v = tr ? src[(long long)c * R + rr] : src[(long long)rr * Csrc + c];
  dst[i] = (f16)v;
}

__global__ void tpn_pe(float* pe) {
  int i = blockIdx.x * blockDim.x + threadIdx.x;
  if (i >= Tq * Hdim) return;
  int t = i / Hdim, d = i - t * Hdim;
  float div = __expf((float)(d & ~1) * (-logf(10000.f) / (float)Hdim));
  float ang = (float)t * div;
  pe[i] = (d & 1) ? __cosf(ang) : __sinf(ang);
}

// QKV f32 (M, ldsrc) -> per-head f16 buffers.
// !tr: dst[(b*NH+h)][t][d]   rows alloc 64, K pad 32   (Q / K)
//  tr: dst[(b*NH+h)][d][t]   16 rows, K pad 64          (V^T)
__global__ void tpn_scatter_head(const float* src, int Tlen, int ldsrc, int col0,
                                 f16* dst, int tr, float scale, int M) {
  int i = blockIdx.x * blockDim.x + threadIdx.x;
  if (i >= M * Hdim) return;
  int m = i >> 7, c = i & 127;
  int b = m / Tlen, t = m - b * Tlen;
  int h = c >> 4, d = c & 15;
  float v = src[(long long)m * ldsrc + col0 + c] * scale;
  if (!tr) dst[((long long)(b * NHh + h) * 64 + t) * 32 + d] = (f16)v;
  else     dst[((long long)(b * NHh + h) * 16 + d) * 64 + t] = (f16)v;
}

__global__ void tpn_attn_softmax(const float* S, f16* P, int Nq, int Nk,
                                 int limit, int nrows) {
  int row = blockIdx.x * 4 + (threadIdx.x >> 5);
  if (row >= nrows) return;
  int lane = threadIdx.x & 31;
  int q = row % Nq, bh = row / Nq;
  const float* s = S + ((long long)bh * 64 + q) * 64;
  f16* p = P + ((long long)bh * 64 + q) * 64;
  int j0 = lane, j1 = lane + 32;
  bool ok0 = (j0 < Nk) && (limit < 0 || j0 <= limit);
  bool ok1 = (j1 < Nk) && (limit < 0 || j1 <= limit);
  float v0 = ok0 ? s[j0] : -1e30f;
  float v1 = ok1 ? s[j1] : -1e30f;
  float m = fmaxf(v0, v1);
  for (int o = 16; o > 0; o >>= 1) m = fmaxf(m, __shfl_xor(m, o, 32));
  float e0 = ok0 ? __expf(v0 - m) : 0.f;
  float e1 = ok1 ? __expf(v1 - m) : 0.f;
  float sum = e0 + e1;
  for (int o = 16; o > 0; o >>= 1) sum += __shfl_xor(sum, o, 32);
  float inv = 1.f / sum;
  p[j0] = (f16)(e0 * inv);
  p[j1] = (f16)(e1 * inv);
}

__global__ __launch_bounds__(128) void tpn_layernorm(const float* x, const float* a,
    const float* gam, const float* bet, float* o32, f16* o16) {
  long long m = blockIdx.x;
  int t = threadIdx.x;
  __shared__ float red[4];
  float v = x[m * 128 + t] + a[m * 128 + t];
  float s = v;
  for (int o = 16; o > 0; o >>= 1) s += __shfl_xor(s, o, 32);
  if ((t & 31) == 0) red[t >> 5] = s;
  __syncthreads();
  float mean = (red[0] + red[1] + red[2] + red[3]) * (1.f / 128.f);
  __syncthreads();
  float dv = v - mean;
  float s2 = dv * dv;
  for (int o = 16; o > 0; o >>= 1) s2 += __shfl_xor(s2, o, 32);
  if ((t & 31) == 0) red[t >> 5] = s2;
  __syncthreads();
  float var = (red[0] + red[1] + red[2] + red[3]) * (1.f / 128.f);
  float y = dv * rsqrtf(var + 1e-5f) * gam[t] + bet[t];
  o32[m * 128 + t] = y;
  o16[m * 128 + t] = (f16)y;
}

__global__ void tpn_add_pe(float* e32, f16* e16, const float* pe, int total) {
  int i = blockIdx.x * blockDim.x + threadIdx.x;
  if (i >= total) return;
  int m = i >> 7, c = i & 127;
  int t = m % Tq;
  float v = e32[i] + pe[t * 128 + c];
  e32[i] = v; e16[i] = (f16)v;
}

__global__ void tpn_init_bos(const float* bos, f16* tgt16) {
  int i = blockIdx.x * blockDim.x + threadIdx.x;
  if (i >= Bb * Hdim) return;
  int b = i >> 7, c = i & 127;
  tgt16[(long long)b * Tq * 128 + c] = (f16)bos[c];
}

__global__ void tpn_extract_q(const float* dd, int step, float* q32, f16* q16) {
  int i = blockIdx.x * blockDim.x + threadIdx.x;
  if (i >= Bb * Hdim) return;
  int b = i >> 7, c = i & 127;
  float v = dd[((long long)(b * Tq) + step) * 128 + c];
  q32[i] = v; q16[i] = (f16)v;
}

__global__ void tpn_ptr_logits(const float* r, const float* qw, const float* vv,
                               const float* mask, float* logits) {
  int row = blockIdx.x * 4 + (threadIdx.x >> 5);   // row = b*32 + n, 2048 total
  int lane = threadIdx.x & 31;
  int b = row >> 5;
  const float* rr = r + (long long)row * 128;
  const float* qq = qw + (long long)b * 128;
  float s = 0.f;
  for (int c = lane; c < 128; c += 32) s += vv[c] * tanhf(rr[c] + qq[c]);
  for (int o = 16; o > 0; o >>= 1) s += __shfl_xor(s, o, 32);
  if (lane == 0) logits[row] = s - 1e7f * mask[row];
}

__global__ void tpn_softmax32(const float* lg, float* pr) {
  int b = blockIdx.x * 4 + (threadIdx.x >> 5);
  int lane = threadIdx.x & 31;
  float v = lg[b * 32 + lane];
  float m = v;
  for (int o = 16; o > 0; o >>= 1) m = fmaxf(m, __shfl_xor(m, o, 32));
  float e = __expf(v - m);
  float s = e;
  for (int o = 16; o > 0; o >>= 1) s += __shfl_xor(s, o, 32);
  pr[b * 32 + lane] = e / s;
}

__global__ void tpn_glimpse_combine(const float* pr, const float* r,
                                    float* q32, f16* q16) {
  int i = blockIdx.x * blockDim.x + threadIdx.x;
  if (i >= Bb * Hdim) return;
  int b = i >> 7, c = i & 127;
  float s = 0.f;
  for (int n = 0; n < 32; ++n)
    s += pr[b * 32 + n] * r[((long long)(b * 32 + n)) * 128 + c];
  q32[i] = s; q16[i] = (f16)s;
}

__global__ void tpn_finalize(const float* logits, const float* xemb, f16* tgt16,
                             float* mask, float* logll, float* out, int step) {
  int b = blockIdx.x;
  int lane = threadIdx.x;
  float v = logits[b * 32 + lane];
  float m = v; int idx = lane;
  for (int o = 16; o > 0; o >>= 1) {
    float om = __shfl_xor(m, o, 32);
    int oi = __shfl_xor(idx, o, 32);
    if (om > m || (om == m && oi < idx)) { m = om; idx = oi; }
  }
  float e = __expf(v - m);
  float s = e;
  for (int o = 16; o > 0; o >>= 1) s += __shfl_xor(s, o, 32);
  float lse = m + __logf(s);
  int pos = idx;                                    // uniform after reduction
  if (lane == 0) {
    out[b * 32 + step] = (float)pos;                // tour (B,N)
    logll[b] += logits[b * 32 + pos] - lse;
    mask[b * 32 + pos] = 1.f;
  }
  for (int c = lane; c < 128; c += 32)
    tgt16[((long long)(b * Tq) + step + 1) * 128 + c] =
        (f16)xemb[((long long)(b * 32 + pos)) * 128 + c];
}

__global__ void tpn_write_logll(const float* logll, float* out) {
  int b = threadIdx.x;
  if (b < Bb) out[Bb * Nn + b] = logll[b];
}

// ----------------------------- host helpers --------------------------------
static void launch_gemm(hipStream_t st,
    const f16* A, long long sA1, long long sA2, int lda,
    const f16* W, long long sW1, long long sW2, int ldw, int Nalloc,
    const float* bias,
    float* C, long long sC1, long long sC2, int ldc,
    f16* C16, long long sD1, long long sD2, int ldc16,
    int M, int N, int K, int NB1, int NB2, int act) {
  GemmArgs g;
  g.A = A; g.sA1 = sA1; g.sA2 = sA2; g.lda = lda;
  g.W = W; g.sW1 = sW1; g.sW2 = sW2; g.ldw = ldw;
  g.bias = bias;
  g.C = C; g.sC1 = sC1; g.sC2 = sC2; g.ldc = ldc;
  g.C16 = C16; g.sD1 = sD1; g.sD2 = sD2; g.ldc16 = ldc16;
  g.M = M; g.N = N; g.K = K; g.NB2 = NB2; g.act = act;
  dim3 grid((M + 63) / 64, (Nalloc + 63) / 64, NB1 * NB2);
  // Nalloc is a multiple of 64 for all operands except V^T (16 rows):
  if (Nalloc % 64 == 0)
    tpn_gemm_wmma<4><<<grid, dim3(128), 0, st>>>(g);
  else
    tpn_gemm_wmma<1><<<grid, dim3(128), 0, st>>>(g);
}

extern "C" void kernel_launch(void* const* d_in, const int* in_sizes, int n_in,
                              void* d_out, int out_size, void* d_ws, size_t ws_size,
                              hipStream_t stream) {
  (void)in_sizes; (void)n_in; (void)out_size; (void)ws_size;
  const float* in_x         = (const float*)d_in[0];
  const float* fc_w         = (const float*)d_in[1];
  const float* fc_b         = (const float*)d_in[2];
  const float* bos          = (const float*)d_in[3];
  const float* demb_w       = (const float*)d_in[4];
  const float* demb_b       = (const float*)d_in[5];
  const float* enc_qkv_w    = (const float*)d_in[6];
  const float* enc_qkv_b    = (const float*)d_in[7];
  const float* enc_out_w    = (const float*)d_in[8];
  const float* enc_out_b    = (const float*)d_in[9];
  const float* enc_ff1_w    = (const float*)d_in[10];
  const float* enc_ff1_b    = (const float*)d_in[11];
  const float* enc_ff2_w    = (const float*)d_in[12];
  const float* enc_ff2_b    = (const float*)d_in[13];
  const float* enc_ln1_g    = (const float*)d_in[14];
  const float* enc_ln1_bb   = (const float*)d_in[15];
  const float* enc_ln2_g    = (const float*)d_in[16];
  const float* enc_ln2_bb   = (const float*)d_in[17];
  const float* dec_sa_qkv_w = (const float*)d_in[18];
  const float* dec_sa_qkv_b = (const float*)d_in[19];
  const float* dec_sa_out_w = (const float*)d_in[20];
  const float* dec_sa_out_b = (const float*)d_in[21];
  const float* dec_ca_qkv_w = (const float*)d_in[22];
  const float* dec_ca_qkv_b = (const float*)d_in[23];
  const float* dec_ca_out_w = (const float*)d_in[24];
  const float* dec_ca_out_b = (const float*)d_in[25];
  const float* dec_ff1_w    = (const float*)d_in[26];
  const float* dec_ff1_b    = (const float*)d_in[27];
  const float* dec_ff2_w    = (const float*)d_in[28];
  const float* dec_ff2_b    = (const float*)d_in[29];
  const float* dec_ln1_g    = (const float*)d_in[30];
  const float* dec_ln1_bb   = (const float*)d_in[31];
  const float* dec_ln2_g    = (const float*)d_in[32];
  const float* dec_ln2_bb   = (const float*)d_in[33];
  const float* dec_ln3_g    = (const float*)d_in[34];
  const float* dec_ln3_bb   = (const float*)d_in[35];
  const float* wref_g       = (const float*)d_in[36];
  const float* wq_g         = (const float*)d_in[37];
  const float* v_g          = (const float*)d_in[38];
  const float* wref_a       = (const float*)d_in[39];
  const float* wq_a         = (const float*)d_in[40];
  const float* v_a          = (const float*)d_in[41];
  float* out = (float*)d_out;

  // ---- workspace bump allocator (deterministic layout every call) ----
  char* base = (char*)d_ws; size_t off = 0;
  auto alloc = [&](size_t bytes) -> char* {
    size_t a = (off + 255) & ~(size_t)255; off = a + bytes; return base + a;
  };
  auto F32 = [&](size_t n) { return (float*)alloc(n * sizeof(float)); };
  auto F16b = [&](size_t n) { return (f16*)alloc(n * sizeof(f16)); };

  float* xemb  = F32((size_t)2048 * 128);
  float* qkv32 = F32((size_t)2112 * 384);
  float* S     = F32((size_t)512 * 64 * 64);
  float* ao32  = F32((size_t)2112 * 128);
  float* a32   = F32((size_t)2112 * 128);
  float* h32   = F32((size_t)2048 * 128);
  float* d32   = F32((size_t)2112 * 128);
  float* emb32 = F32((size_t)2112 * 128);
  float* rg32  = F32((size_t)2048 * 128);
  float* ra32  = F32((size_t)2048 * 128);
  float* q32   = F32(64 * 128);
  float* qw32  = F32(64 * 128);
  float* gl    = F32(64 * 32);
  float* gp    = F32(64 * 32);
  float* la    = F32(64 * 32);
  float* mask  = F32(64 * 32);
  float* logll = F32(64);
  float* pe    = F32(33 * 128);

  f16* fcw16   = F16b(128 * 32);
  f16* demb16  = F16b(128 * 128);
  f16* eqkv16  = F16b((size_t)2 * 384 * 128);
  f16* eout16  = F16b((size_t)2 * 128 * 128);
  f16* eff1_16 = F16b((size_t)2 * 2048 * 128);
  f16* eff2_16 = F16b((size_t)2 * 128 * 2048);
  f16* saqkv16 = F16b((size_t)2 * 384 * 128);
  f16* saout16 = F16b((size_t)2 * 128 * 128);
  f16* caqkv16 = F16b((size_t)2 * 384 * 128);
  f16* caout16 = F16b((size_t)2 * 128 * 128);
  f16* dff1_16 = F16b((size_t)2 * 2048 * 128);
  f16* dff2_16 = F16b((size_t)2 * 128 * 2048);
  f16* wrgT16  = F16b(128 * 128);
  f16* wqgT16  = F16b(128 * 128);
  f16* wraT16  = F16b(128 * 128);
  f16* wqaT16  = F16b(128 * 128);

  f16* x16   = F16b((size_t)2048 * 32);
  f16* h16   = F16b((size_t)2048 * 128);
  f16* tgt16 = F16b((size_t)2112 * 128);
  f16* emb16 = F16b((size_t)2112 * 128);
  f16* d16   = F16b((size_t)2112 * 128);
  f16* ao16  = F16b((size_t)2112 * 128);
  f16* ff16  = F16b((size_t)2112 * 2048);
  f16* q16   = F16b(64 * 128);
  f16* Qh    = F16b((size_t)512 * 64 * 32);
  f16* Kh    = F16b((size_t)512 * 64 * 32);
  f16* Vt    = F16b((size_t)512 * 16 * 64);
  f16* Kc0   = F16b((size_t)512 * 64 * 32);
  f16* Kc1   = F16b((size_t)512 * 64 * 32);
  f16* Vtc0  = F16b((size_t)512 * 16 * 64);
  f16* Vtc1  = F16b((size_t)512 * 16 * 64);
  f16* P16   = F16b((size_t)512 * 64 * 64);
  f16* KcL[2]  = {Kc0, Kc1};
  f16* VtcL[2] = {Vtc0, Vtc1};

  // batched-attention strides
  const long long QS_h = 64 * 32, QS_b = NHh * QS_h;   // Q/K per-head
  const long long VT_h = 16 * 64, VT_b = NHh * VT_h;   // V^T per-head
  const long long S_h  = 64 * 64, S_b  = NHh * S_h;    // scores / probs

  // ---- per-launch zero init (padding, mask, logll, tgt) ----
  hipMemsetAsync(tgt16, 0, (size_t)2112 * 128 * 2, stream);
  hipMemsetAsync(Qh,   0, (size_t)512 * 64 * 32 * 2, stream);
  hipMemsetAsync(Kh,   0, (size_t)512 * 64 * 32 * 2, stream);
  hipMemsetAsync(Vt,   0, (size_t)512 * 16 * 64 * 2, stream);
  hipMemsetAsync(Kc0,  0, (size_t)512 * 64 * 32 * 2, stream);
  hipMemsetAsync(Kc1,  0, (size_t)512 * 64 * 32 * 2, stream);
  hipMemsetAsync(Vtc0, 0, (size_t)512 * 16 * 64 * 2, stream);
  hipMemsetAsync(Vtc1, 0, (size_t)512 * 16 * 64 * 2, stream);
  hipMemsetAsync(mask, 0, 64 * 32 * 4, stream);
  hipMemsetAsync(logll, 0, 64 * 4, stream);

  tpn_pe<<<(33 * 128 + 255) / 256, 256, 0, stream>>>(pe);

  auto cvt = [&](const float* src, f16* dst, int R, int C, int ld, int tr) {
    int n = R * ld;
    tpn_cvt_f32_f16<<<(n + 255) / 256, 256, 0, stream>>>(src, dst, R, C, ld, tr);
  };
  cvt(fc_w, fcw16, 128, 2, 32, 0);
  cvt(in_x, x16, 2048, 2, 32, 0);
  cvt(demb_w, demb16, 128, 128, 128, 0);
  for (int l = 0; l < 2; ++l) {
    cvt(enc_qkv_w + (size_t)l * 384 * 128,  eqkv16 + (size_t)l * 384 * 128,  384, 128, 128, 0);
    cvt(enc_out_w + (size_t)l * 128 * 128,  eout16 + (size_t)l * 128 * 128,  128, 128, 128, 0);
    cvt(enc_ff1_w + (size_t)l * 2048 * 128, eff1_16 + (size_t)l * 2048 * 128, 2048, 128, 128, 0);
    cvt(enc_ff2_w + (size_t)l * 128 * 2048, eff2_16 + (size_t)l * 128 * 2048, 128, 2048, 2048, 0);
    cvt(dec_sa_qkv_w + (size_t)l * 384 * 128, saqkv16 + (size_t)l * 384 * 128, 384, 128, 128, 0);
    cvt(dec_sa_out_w + (size_t)l * 128 * 128, saout16 + (size_t)l * 128 * 128, 128, 128, 128, 0);
    cvt(dec_ca_qkv_w + (size_t)l * 384 * 128, caqkv16 + (size_t)l * 384 * 128, 384, 128, 128, 0);
    cvt(dec_ca_out_w + (size_t)l * 128 * 128, caout16 + (size_t)l * 128 * 128, 128, 128, 128, 0);
    cvt(dec_ff1_w + (size_t)l * 2048 * 128, dff1_16 + (size_t)l * 2048 * 128, 2048, 128, 128, 0);
    cvt(dec_ff2_w + (size_t)l * 128 * 2048, dff2_16 + (size_t)l * 128 * 2048, 128, 2048, 2048, 0);
  }
  cvt(wref_g, wrgT16, 128, 128, 128, 1);   // r = mem @ wref  -> W = wref^T
  cvt(wq_g,  wqgT16, 128, 128, 128, 1);
  cvt(wref_a, wraT16, 128, 128, 128, 1);
  cvt(wq_a,  wqaT16, 128, 128, 128, 1);
  tpn_init_bos<<<(64 * 128 + 255) / 256, 256, 0, stream>>>(bos, tgt16);

  // ============================ ENCODER ============================
  launch_gemm(stream, x16, 0, 0, 32, fcw16, 0, 0, 32, 128, fc_b,
              xemb, 0, 0, 128, h16, 0, 0, 128, 2048, 128, 32, 1, 1, 0);
  const float* hx = xemb;
  for (int l = 0; l < 2; ++l) {
    launch_gemm(stream, h16, 0, 0, 128, eqkv16 + (size_t)l * 384 * 128, 0, 0, 128, 384,
                enc_qkv_b + l * 384, qkv32, 0, 0, 384, nullptr, 0, 0, 0,
                2048, 384, 128, 1, 1, 0);
    tpn_scatter_head<<<(2048 * 128 + 255) / 256, 256, 0, stream>>>(qkv32, 32, 384, 0,   Qh, 0, 0.25f, 2048);
    tpn_scatter_head<<<(2048 * 128 + 255) / 256, 256, 0, stream>>>(qkv32, 32, 384, 128, Kh, 0, 1.f,   2048);
    tpn_scatter_head<<<(2048 * 128 + 255) / 256, 256, 0, stream>>>(qkv32, 32, 384, 256, Vt, 1, 1.f,   2048);
    launch_gemm(stream, Qh, QS_b, QS_h, 32, Kh, QS_b, QS_h, 32, 64, nullptr,
                S, S_b, S_h, 64, nullptr, 0, 0, 0, 32, 32, 32, 64, 8, 0);
    tpn_attn_softmax<<<(64 * 8 * 32 + 3) / 4, 128, 0, stream>>>(S, P16, 32, 32, -1, 64 * 8 * 32);
    launch_gemm(stream, P16, S_b, S_h, 64, Vt, VT_b, VT_h, 64, 16, nullptr,
                ao32, (long long)32 * 128, 16, 128, ao16, (long long)32 * 128, 16, 128,
                32, 16, 64, 64, 8, 0);
    launch_gemm(stream, ao16, 0, 0, 128, eout16 + (size_t)l * 128 * 128, 0, 0, 128, 128,
                enc_out_b + l * 128, a32, 0, 0, 128, nullptr, 0, 0, 0, 2048, 128, 128, 1, 1, 0);
    tpn_layernorm<<<2048, 128, 0, stream>>>(hx, a32, enc_ln1_g + l * 128, enc_ln1_bb + l * 128, h32, h16);
    hx = h32;
    launch_gemm(stream, h16, 0, 0, 128, eff1_16 + (size_t)l * 2048 * 128, 0, 0, 128, 2048,
                enc_ff1_b + l * 2048, nullptr, 0, 0, 0, ff16, 0, 0, 2048, 2048, 2048, 128, 1, 1, 1);
    launch_gemm(stream, ff16, 0, 0, 2048, eff2_16 + (size_t)l * 128 * 2048, 0, 0, 2048, 128,
                enc_ff2_b + l * 128, a32, 0, 0, 128, nullptr, 0, 0, 0, 2048, 128, 2048, 1, 1, 0);
    tpn_layernorm<<<2048, 128, 0, stream>>>(h32, a32, enc_ln2_g + l * 128, enc_ln2_bb + l * 128, h32, h16);
  }

  // ---- memory-dependent precompute (cross-attn K/V, pointer r-matrices) ----
  for (int l = 0; l < 2; ++l) {
    launch_gemm(stream, h16, 0, 0, 128,
                caqkv16 + (size_t)l * 384 * 128 + (size_t)128 * 128, 0, 0, 128, 256,
                dec_ca_qkv_b + l * 384 + 128, qkv32, 0, 0, 256, nullptr, 0, 0, 0,
                2048, 256, 128, 1, 1, 0);
    tpn_scatter_head<<<(2048 * 128 + 255) / 256, 256, 0, stream>>>(qkv32, 32, 256, 0,   KcL[l],  0, 1.f, 2048);
    tpn_scatter_head<<<(2048 * 128 + 255) / 256, 256, 0, stream>>>(qkv32, 32, 256, 128, VtcL[l], 1, 1.f, 2048);
  }
  launch_gemm(stream, h16, 0, 0, 128, wrgT16, 0, 0, 128, 128, nullptr,
              rg32, 0, 0, 128, nullptr, 0, 0, 0, 2048, 128, 128, 1, 1, 0);
  launch_gemm(stream, h16, 0, 0, 128, wraT16, 0, 0, 128, 128, nullptr,
              ra32, 0, 0, 128, nullptr, 0, 0, 0, 2048, 128, 128, 1, 1, 0);

  // ============================ DECODER LOOP ============================
  for (int i = 0; i < Nn; ++i) {
    launch_gemm(stream, tgt16, 0, 0, 128, demb16, 0, 0, 128, 128, demb_b,
                emb32, 0, 0, 128, emb16, 0, 0, 128, 2112, 128, 128, 1, 1, 1);
    tpn_add_pe<<<(2112 * 128 + 255) / 256, 256, 0, stream>>>(emb32, emb16, pe, 2112 * 128);
    const float* dx = emb32;
    const f16* dr = emb16;
    for (int l = 0; l < 2; ++l) {
      // ---- self attention (keys masked to <= i) ----
      launch_gemm(stream, dr, 0, 0, 128, saqkv16 + (size_t)l * 384 * 128, 0, 0, 128, 384,
                  dec_sa_qkv_b + l * 384, qkv32, 0, 0, 384, nullptr, 0, 0, 0,
                  2112, 384, 128, 1, 1, 0);
      tpn_scatter_head<<<(2112 * 128 + 255) / 256, 256, 0, stream>>>(qkv32, 33, 384, 0,   Qh, 0, 0.25f, 2112);
      tpn_scatter_head<<<(2112 * 128 + 255) / 256, 256, 0, stream>>>(qkv32, 33, 384, 128, Kh, 0, 1.f,   2112);
      tpn_scatter_head<<<(2112 * 128 + 255) / 256, 256, 0, stream>>>(qkv32, 33, 384, 256, Vt, 1, 1.f,   2112);
      launch_gemm(stream, Qh, QS_b, QS_h, 32, Kh, QS_b, QS_h, 32, 64, nullptr,
                  S, S_b, S_h, 64, nullptr, 0, 0, 0, 33, 33, 32, 64, 8, 0);
      tpn_attn_softmax<<<(64 * 8 * 33 + 3) / 4, 128, 0, stream>>>(S, P16, 33, 33, i, 64 * 8 * 33);
      launch_gemm(stream, P16, S_b, S_h, 64, Vt, VT_b, VT_h, 64, 16, nullptr,
                  ao32, (long long)33 * 128, 16, 128, ao16, (long long)33 * 128, 16, 128,
                  33, 16, 64, 64, 8, 0);
      launch_gemm(stream, ao16, 0, 0, 128, saout16 + (size_t)l * 128 * 128, 0, 0, 128, 128,
                  dec_sa_out_b + l * 128, a32, 0, 0, 128, nullptr, 0, 0, 0, 2112, 128, 128, 1, 1, 0);
      tpn_layernorm<<<2112, 128, 0, stream>>>(dx, a32, dec_ln1_g + l * 128, dec_ln1_bb + l * 128, d32, d16);
      dx = d32; dr = d16;
      // ---- cross attention (K/V precomputed from memory) ----
      launch_gemm(stream, dr, 0, 0, 128, caqkv16 + (size_t)l * 384 * 128, 0, 0, 128, 128,
                  dec_ca_qkv_b + l * 384, qkv32, 0, 0, 128, nullptr, 0, 0, 0,
                  2112, 128, 128, 1, 1, 0);
      tpn_scatter_head<<<(2112 * 128 + 255) / 256, 256, 0, stream>>>(qkv32, 33, 128, 0, Qh, 0, 0.25f, 2112);
      launch_gemm(stream, Qh, QS_b, QS_h, 32, KcL[l], QS_b, QS_h, 32, 64, nullptr,
                  S, S_b, S_h, 64, nullptr, 0, 0, 0, 33, 32, 32, 64, 8, 0);
      tpn_attn_softmax<<<(64 * 8 * 33 + 3) / 4, 128, 0, stream>>>(S, P16, 33, 32, -1, 64 * 8 * 33);
      launch_gemm(stream, P16, S_b, S_h, 64, VtcL[l], VT_b, VT_h, 64, 16, nullptr,
                  ao32, (long long)33 * 128, 16, 128, ao16, (long long)33 * 128, 16, 128,
                  33, 16, 64, 64, 8, 0);
      launch_gemm(stream, ao16, 0, 0, 128, caout16 + (size_t)l * 128 * 128, 0, 0, 128, 128,
                  dec_ca_out_b + l * 128, a32, 0, 0, 128, nullptr, 0, 0, 0, 2112, 128, 128, 1, 1, 0);
      tpn_layernorm<<<2112, 128, 0, stream>>>(d32, a32, dec_ln2_g + l * 128, dec_ln2_bb + l * 128, d32, d16);
      // ---- feed-forward ----
      launch_gemm(stream, d16, 0, 0, 128, dff1_16 + (size_t)l * 2048 * 128, 0, 0, 128, 2048,
                  dec_ff1_b + l * 2048, nullptr, 0, 0, 0, ff16, 0, 0, 2048, 2112, 2048, 128, 1, 1, 1);
      launch_gemm(stream, ff16, 0, 0, 2048, dff2_16 + (size_t)l * 128 * 2048, 0, 0, 2048, 128,
                  dec_ff2_b + l * 128, a32, 0, 0, 128, nullptr, 0, 0, 0, 2112, 128, 2048, 1, 1, 0);
      tpn_layernorm<<<2112, 128, 0, stream>>>(d32, a32, dec_ln3_g + l * 128, dec_ln3_bb + l * 128, d32, d16);
    }
    tpn_extract_q<<<(64 * 128 + 255) / 256, 256, 0, stream>>>(d32, i, q32, q16);
    // glimpse (NG = 1)
    launch_gemm(stream, q16, 0, 0, 128, wqgT16, 0, 0, 128, 128, nullptr,
                qw32, 0, 0, 128, nullptr, 0, 0, 0, 64, 128, 128, 1, 1, 0);
    tpn_ptr_logits<<<512, 128, 0, stream>>>(rg32, qw32, v_g, mask, gl);
    tpn_softmax32<<<16, 128, 0, stream>>>(gl, gp);
    tpn_glimpse_combine<<<(64 * 128 + 255) / 256, 256, 0, stream>>>(gp, rg32, q32, q16);
    // pointer attention + step update
    launch_gemm(stream, q16, 0, 0, 128, wqaT16, 0, 0, 128, 128, nullptr,
                qw32, 0, 0, 128, nullptr, 0, 0, 0, 64, 128, 128, 1, 1, 0);
    tpn_ptr_logits<<<512, 128, 0, stream>>>(ra32, qw32, v_a, mask, la);
    tpn_finalize<<<64, 32, 0, stream>>>(la, xemb, tgt16, mask, logll, out, i);
  }
  tpn_write_logll<<<1, 64, 0, stream>>>(logll, out);
}
